// FixedSlotAttentionMultiHeadProb_56315611185661
// MI455X (gfx1250) — matrix operand order repaired and checked by reading, not compiled, not run on hardware
//
#include <hip/hip_runtime.h>
#include <hip/hip_bf16.h>

// ---------------- problem constants ----------------
constexpr int kB = 16, kN = 4096, kD = 64, kK = 8, kH = 4, kDH = 16, kHID = 256, kITERS = 3;
constexpr float kEPS = 1e-5f;

typedef __attribute__((ext_vector_type(16))) _Float16 v16h;
typedef __attribute__((ext_vector_type(8)))  _Float16 v8h;
typedef __attribute__((ext_vector_type(8)))  float    v8f;

// ---------------- init: slots, sigma, mixing, qdiag ----------------
__global__ void init_kernel(const float* __restrict__ mu, const float* __restrict__ logsig,
                            const float* __restrict__ mix0, const float* __restrict__ to_q,
                            const float* __restrict__ noise0,
                            float* __restrict__ slots, float* __restrict__ sigma,
                            float* __restrict__ mixing, float* __restrict__ qdiag) {
  int t = blockIdx.x * 256 + threadIdx.x;
  if (t < kB * kK * kD) {
    int d = t % kD;
    int k = (t / kD) % kK;
    float sg = __expf(logsig[k * kD + d]);
    sigma[t] = sg;
    slots[t] = mu[k * kD + d] + sg * noise0[t];
  }
  if (t < kB * kK) mixing[t] = mix0[t % kK];
  if (t < kK * kD) {
    int k = t / kD, d = t % kD;
    qdiag[t] = to_q[((size_t)k * kD + d) * kD + d];   // diagonal of to_queries
  }
}

// ---------------- weights: transpose + f16 convert ----------------
__global__ void wconv_kernel(const float* __restrict__ wk, const float* __restrict__ wv,
                             _Float16* __restrict__ wkT, _Float16* __restrict__ wvT) {
  int t = blockIdx.x * 256 + threadIdx.x;          // over K*D*D
  if (t < kK * kD * kD) {
    int d = t % kD;
    int e = (t / kD) % kD;
    int k = t / (kD * kD);
    size_t src = ((size_t)k * kD + e) * kD + d;    // [k][e][d]
    size_t dst = ((size_t)k * kD + d) * kD + e;    // [k][d][e]
    wkT[dst] = (_Float16)wk[src];
    wvT[dst] = (_Float16)wv[src];
  }
}

// ---------------- LayerNorm(embeddings) -> f16 ----------------
__global__ __launch_bounds__(256) void ln_kernel(const float* __restrict__ emb,
                                                 const float* __restrict__ g,
                                                 const float* __restrict__ bb,
                                                 _Float16* __restrict__ xh) {
  const int lane = threadIdx.x & 31;
  const int wave = threadIdx.x >> 5;
  const size_t row = (size_t)blockIdx.x * 8 + wave;        // B*N rows
  const float* p = emb + row * kD + lane * 2;
  float2 v = *(const float2*)p;
  float s = v.x + v.y, ss = v.x * v.x + v.y * v.y;
  #pragma unroll
  for (int m = 16; m >= 1; m >>= 1) { s += __shfl_xor(s, m, 32); ss += __shfl_xor(ss, m, 32); }
  float mean = s * (1.f / 64.f);
  float var  = ss * (1.f / 64.f) - mean * mean;
  float r = rsqrtf(var + kEPS);
  _Float16 h0 = (_Float16)((v.x - mean) * r * g[lane * 2]     + bb[lane * 2]);
  _Float16 h1 = (_Float16)((v.y - mean) * r * g[lane * 2 + 1] + bb[lane * 2 + 1]);
  xh[row * kD + lane * 2]     = h0;
  xh[row * kD + lane * 2 + 1] = h1;
}

// ---------------- WMMA GEMM: keys & values projections ----------------
// keys[b,k,n,:] = x_ln[b,n,:] @ to_keys[k]   (f16 in, f32 accum, f16 out)
__global__ __launch_bounds__(128) void gemm_kernel(const _Float16* __restrict__ xh,
                                                   const _Float16* __restrict__ wkT,
                                                   const _Float16* __restrict__ wvT,
                                                   _Float16* __restrict__ keys,
                                                   _Float16* __restrict__ vals) {
  const int bk   = blockIdx.x;            // b*K + k
  const int b    = bk >> 3;
  const int k    = bk & 7;
  const int wave = threadIdx.x >> 5;
  const int lane = threadIdx.x & 31;
  const int half = lane >> 4;
  const int r    = lane & 15;
  const int m0   = (blockIdx.y * 4 + wave) * 16;

  // A fragments (ISA layout: lane<16 holds K {0..7,16..23}; lane>=16 holds {8..15,24..31})
  const _Float16* xrow = xh + ((size_t)b * kN + m0 + r) * kD;
  union AF { v16h v; v8h h[2]; };
  AF a0, a1;
  a0.h[0] = *(const v8h*)(xrow + half * 8);
  a0.h[1] = *(const v8h*)(xrow + 16 + half * 8);
  a1.h[0] = *(const v8h*)(xrow + 32 + half * 8);
  a1.h[1] = *(const v8h*)(xrow + 48 + half * 8);

  #pragma unroll
  for (int dt = 0; dt < 4; ++dt) {
    const int dcol = dt * 16 + r;
    // B fragments from transposed weights [k][d][e]: lane<16 -> K rows {0..7,16..23}, lane>=16 -> {8..15,24..31}
    const _Float16* wkrow = wkT + ((size_t)k * kD + dcol) * kD + half * 8;
    const _Float16* wvrow = wvT + ((size_t)k * kD + dcol) * kD + half * 8;

    union AF bk0, bk1;
    bk0.h[0] = *(const v8h*)(wkrow);      bk0.h[1] = *(const v8h*)(wkrow + 16);
    bk1.h[0] = *(const v8h*)(wkrow + 32); bk1.h[1] = *(const v8h*)(wkrow + 48);
    v8f c = {};
    c = __builtin_amdgcn_wmma_f32_16x16x32_f16(false, a0.v, false, bk0.v, (short)0, c, false, false);
    c = __builtin_amdgcn_wmma_f32_16x16x32_f16(false, a1.v, false, bk1.v, (short)0, c, false, false);
    _Float16* kout = keys + (((size_t)bk * kN + m0) * kD) + dcol;
    #pragma unroll
    for (int j = 0; j < 8; ++j) kout[(size_t)(j + 8 * half) * kD] = (_Float16)c[j];

    union AF bv0, bv1;
    bv0.h[0] = *(const v8h*)(wvrow);      bv0.h[1] = *(const v8h*)(wvrow + 16);
    bv1.h[0] = *(const v8h*)(wvrow + 32); bv1.h[1] = *(const v8h*)(wvrow + 48);
    v8f cv = {};
    cv = __builtin_amdgcn_wmma_f32_16x16x32_f16(false, a0.v, false, bv0.v, (short)0, cv, false, false);
    cv = __builtin_amdgcn_wmma_f32_16x16x32_f16(false, a1.v, false, bv1.v, (short)0, cv, false, false);
    _Float16* vout = vals + (((size_t)bk * kN + m0) * kD) + dcol;
    #pragma unroll
    for (int j = 0; j < 8; ++j) vout[(size_t)(j + 8 * half) * kD] = (_Float16)cv[j];
  }
}

// ---------------- gll: gaussian log-likelihood logits ----------------
// gll[b,k,n,h] = c0[h] + sum_dh( k*qw - 0.5*k^2*w )
__global__ __launch_bounds__(256) void gll_kernel(const _Float16* __restrict__ keys,
                                                  const float* __restrict__ slots,
                                                  const float* __restrict__ sigma,
                                                  const float* __restrict__ mixing,
                                                  const float* __restrict__ qdiag,
                                                  const float* __restrict__ lsg,
                                                  const float* __restrict__ lsb,
                                                  float* __restrict__ gll) {
  __shared__ float s_sl[64], s_qw[64], s_w[64], s_t1[64], s_c0[4], s_mv[2];
  const int bk = blockIdx.x;
  const int k  = bk & 7;
  const int tid = threadIdx.x;
  if (tid < 32) {                        // wave-parallel LayerNorm stats
    float2 v = *(const float2*)(slots + bk * 64 + tid * 2);
    s_sl[tid * 2] = v.x; s_sl[tid * 2 + 1] = v.y;
    float s = v.x + v.y, ss = v.x * v.x + v.y * v.y;
    #pragma unroll
    for (int m = 16; m >= 1; m >>= 1) { s += __shfl_xor(s, m, 32); ss += __shfl_xor(ss, m, 32); }
    if (tid == 0) {
      float mean = s * (1.f / 64.f);
      s_mv[0] = mean;
      s_mv[1] = rsqrtf(ss * (1.f / 64.f) - mean * mean + kEPS);
    }
  }
  __syncthreads();
  if (tid < 64) {
    float m = s_mv[0], r = s_mv[1];
    float sl = (s_sl[tid] - m) * r * lsg[tid] + lsb[tid];
    float q  = sl * qdiag[k * 64 + tid];
    float sg = sigma[bk * 64 + tid];
    float w  = 1.f / (sg * sg + kEPS);
    s_qw[tid] = q * w;
    s_w[tid]  = w;
    s_t1[tid] = -__logf(fmaxf(sg, kEPS)) - 0.5f * q * q * w;   // log_scale - 0.5*q^2*w
  }
  __syncthreads();
  if (tid < 4) {
    const float logpi = -0.5f * 64.f * 1.8378770664093453f;    // -0.5*D*log(2*pi)
    float c = __logf(mixing[bk] + kEPS) + 16.f * logpi;
    for (int dh = 0; dh < 16; ++dh) c += s_t1[tid * 16 + dh];
    s_c0[tid] = c;
  }
  __syncthreads();
  const int nbase = blockIdx.y * 256;
  #pragma unroll
  for (int it = 0; it < 4; ++it) {
    int idx = it * 256 + tid;               // 1024 (n,h) pairs per block
    int n = nbase + (idx >> 2);
    int h = idx & 3;
    const _Float16* kp = keys + ((size_t)bk * kN + n) * 64 + h * 16;
    v8h k0 = *(const v8h*)kp;
    v8h k1 = *(const v8h*)(kp + 8);
    float acc = s_c0[h];
    #pragma unroll
    for (int j = 0; j < 8; ++j) {
      float kf = (float)k0[j]; int d = h * 16 + j;
      acc += kf * s_qw[d] - 0.5f * kf * kf * s_w[d];
      float kg = (float)k1[j]; int d2 = h * 16 + 8 + j;
      acc += kg * s_qw[d2] - 0.5f * kg * kg * s_w[d2];
    }
    gll[((size_t)bk * kN + n) * 4 + h] = acc;
  }
}

// ---------------- softmax over K + single-pass moment accumulation ----------------
// Banked LDS accumulators: [k][h][49 values][8 copies], per-(k,h) cell padded to
// 400 floats. Within one ds_add_f32 a wave's lanes hit banks h*16+copy (32 distinct
// banks) -> conflict-free LDS atomics.
constexpr int kCell = 400;   // 49*8 = 392 padded to 400 (400 mod 64 == 16)

__global__ __launch_bounds__(256) void attn_kernel(const float* __restrict__ gll,
                                                   const _Float16* __restrict__ keys,
                                                   const _Float16* __restrict__ vals,
                                                   float* __restrict__ accS0, float* __restrict__ accS1,
                                                   float* __restrict__ accS2, float* __restrict__ accV1,
                                                   float* __restrict__ attn_out, int write_attn) {
  __shared__ float acc[kK * kH * kCell];   // 12800 floats = 50 KB
  const int tid  = threadIdx.x;
  const int copy = (tid >> 2) & 7;
  for (int i = tid; i < kK * kH * kCell; i += 256) acc[i] = 0.f;
  __syncthreads();

  const int b = blockIdx.y;
  const int n = blockIdx.x * 64 + (tid >> 2);
  const int h = tid & 3;
  float g[8];
  float mx = -3.4e38f;
  #pragma unroll
  for (int k = 0; k < 8; ++k) {
    g[k] = gll[(((size_t)(b * 8 + k)) * kN + n) * 4 + h];
    mx = fmaxf(mx, g[k]);
  }
  float s = 0.f;
  #pragma unroll
  for (int k = 0; k < 8; ++k) { g[k] = __expf(g[k] - mx); s += g[k]; }
  float rs = 1.f / s;

  #pragma unroll
  for (int k = 0; k < 8; ++k) {
    float a = g[k] * rs + kEPS;
    if (write_attn) attn_out[(((size_t)(b * 8 + k)) * kN + n) * 4 + h] = a;
    const _Float16* kp = keys + (((size_t)(b * 8 + k)) * kN + n) * 64 + h * 16;
    const _Float16* vp = vals + (((size_t)(b * 8 + k)) * kN + n) * 64 + h * 16;
    v8h k0 = *(const v8h*)kp;
    v8h k1 = *(const v8h*)(kp + 8);
    v8h v0 = *(const v8h*)vp;
    v8h v1 = *(const v8h*)(vp + 8);
    float* base = acc + (k * 4 + h) * kCell + copy;
    atomicAdd(base + 48 * 8, a);                       // S0
    #pragma unroll
    for (int j = 0; j < 8; ++j) {
      float kf = (float)k0[j];
      atomicAdd(base + j * 8,        a * kf);          // S1[j]
      atomicAdd(base + (16 + j) * 8, a * kf * kf);     // S2[j]
      atomicAdd(base + (32 + j) * 8, a * (float)v0[j]);// V1[j]
      float kg = (float)k1[j];
      atomicAdd(base + (8 + j) * 8,  a * kg);          // S1[8+j]
      atomicAdd(base + (24 + j) * 8, a * kg * kg);     // S2[8+j]
      atomicAdd(base + (40 + j) * 8, a * (float)v1[j]);// V1[8+j]
    }
  }
  __syncthreads();

  // flush: sum 8 copies per cell, one global atomic each
  for (int i = tid; i < kK * kH * 49; i += 256) {
    int kh = i / 49, v = i % 49;
    const float* p = acc + kh * kCell + v * 8;
    float sum = p[0] + p[1] + p[2] + p[3] + p[4] + p[5] + p[6] + p[7];
    if (v == 48)      atomicAdd(&accS0[b * 32  + kh], sum);
    else if (v < 16)  atomicAdd(&accS1[b * 512 + kh * 16 + v], sum);
    else if (v < 32)  atomicAdd(&accS2[b * 512 + kh * 16 + (v - 16)], sum);
    else              atomicAdd(&accV1[b * 512 + kh * 16 + (v - 32)], sum);
  }
}

// ---------------- finalize: upd, sigma, mixing, GRU, FFN ----------------
__global__ __launch_bounds__(256) void finalize_kernel(
    const float* __restrict__ accS0, const float* __restrict__ accS1,
    const float* __restrict__ accS2, const float* __restrict__ accV1,
    float* __restrict__ slots, float* __restrict__ sigma, float* __restrict__ mixing,
    const float* __restrict__ w_ih, const float* __restrict__ w_hh,
    const float* __restrict__ b_ih, const float* __restrict__ b_hh,
    const float* __restrict__ w1, const float* __restrict__ b1,
    const float* __restrict__ w2, const float* __restrict__ b2,
    const float* __restrict__ lfg, const float* __restrict__ lfb) {
  __shared__ float s_prev[64], s_upd[64], s_xg[192], s_hg[192], s_new[64], s_hid[256], s_mv[2];
  const int bk = blockIdx.x;
  const int tid = threadIdx.x;
  const float Nk = accS0[bk * 4 + 0] + accS0[bk * 4 + 1] + accS0[bk * 4 + 2] + accS0[bk * 4 + 3];
  const float rN = 1.f / Nk + kEPS;
  if (tid < 64) {
    s_prev[tid] = slots[bk * 64 + tid];
    int h = tid >> 4;
    float v1  = accV1[bk * 64 + tid];
    float s1  = accS1[bk * 64 + tid];
    float s2  = accS2[bk * 64 + tid];
    float s0h = accS0[bk * 4 + h];
    float upd = rN * v1;
    float sig2 = s2 - 2.f * upd * s1 + upd * upd * s0h;   // sum a*(k-upd)^2
    sigma[bk * 64 + tid] = sqrtf(fmaxf(sig2, 0.f) / Nk) + kEPS;
    s_upd[tid] = upd;
  }
  if (tid == 0) mixing[bk] = Nk / (float)kN;
  __syncthreads();
  if (tid < 192) {        // GRU gates (torch order r,z,n)
    float xg = b_ih[tid], hg = b_hh[tid];
    const float* wi = w_ih + (size_t)tid * 64;
    const float* wh = w_hh + (size_t)tid * 64;
    for (int d = 0; d < 64; ++d) { xg += s_upd[d] * wi[d]; hg += s_prev[d] * wh[d]; }
    s_xg[tid] = xg; s_hg[tid] = hg;
  }
  __syncthreads();
  if (tid < 64) {
    float r  = 1.f / (1.f + __expf(-(s_xg[tid] + s_hg[tid])));
    float z  = 1.f / (1.f + __expf(-(s_xg[64 + tid] + s_hg[64 + tid])));
    float nn = tanhf(s_xg[128 + tid] + r * s_hg[128 + tid]);
    s_new[tid] = (1.f - z) * nn + z * s_prev[tid];
  }
  __syncthreads();
  if (tid == 0) {
    float s = 0.f, ss = 0.f;
    for (int d = 0; d < 64; ++d) { float v = s_new[d]; s += v; ss += v * v; }
    float m = s * (1.f / 64.f);
    s_mv[0] = m;
    s_mv[1] = rsqrtf(ss * (1.f / 64.f) - m * m + kEPS);
  }
  __syncthreads();
  {                       // FFN hidden: relu(LN(slots) @ w1.T + b1)
    float m = s_mv[0], r = s_mv[1];
    float acc1 = b1[tid];
    const float* w = w1 + (size_t)tid * 64;
    for (int d = 0; d < 64; ++d) {
      float pre = (s_new[d] - m) * r * lfg[d] + lfb[d];
      acc1 += pre * w[d];
    }
    s_hid[tid] = fmaxf(acc1, 0.f);
  }
  __syncthreads();
  if (tid < 64) {
    float acc2 = s_new[tid] + b2[tid];
    const float* w = w2 + (size_t)tid * 256;
    for (int hh = 0; hh < 256; ++hh) acc2 += s_hid[hh] * w[hh];
    slots[bk * 64 + tid] = acc2;
  }
}

// ---------------- output: slots + sigma*noise_final ----------------
__global__ void out_kernel(const float* __restrict__ slots, const float* __restrict__ sigma,
                           const float* __restrict__ noisef, float* __restrict__ out) {
  int t = blockIdx.x * 256 + threadIdx.x;
  if (t < kB * kK * kD) out[t] = slots[t] + sigma[t] * noisef[t];
}

// ---------------- host launcher ----------------
extern "C" void kernel_launch(void* const* d_in, const int* in_sizes, int n_in,
                              void* d_out, int out_size, void* d_ws, size_t ws_size,
                              hipStream_t stream) {
  const float* embeddings  = (const float*)d_in[0];
  const float* slots_mu    = (const float*)d_in[1];
  const float* slots_lsig  = (const float*)d_in[2];
  const float* mixing_init = (const float*)d_in[3];
  const float* to_keys     = (const float*)d_in[4];
  const float* to_queries  = (const float*)d_in[5];
  const float* to_values   = (const float*)d_in[6];
  const float* w_ih        = (const float*)d_in[7];
  const float* w_hh        = (const float*)d_in[8];
  const float* b_ih        = (const float*)d_in[9];
  const float* b_hh        = (const float*)d_in[10];
  const float* w1          = (const float*)d_in[11];
  const float* b1          = (const float*)d_in[12];
  const float* w2          = (const float*)d_in[13];
  const float* b2          = (const float*)d_in[14];
  const float* ln_in_g     = (const float*)d_in[15];
  const float* ln_in_b     = (const float*)d_in[16];
  const float* ln_sl_g     = (const float*)d_in[17];
  const float* ln_sl_b     = (const float*)d_in[18];
  const float* ln_ff_g     = (const float*)d_in[19];
  const float* ln_ff_b     = (const float*)d_in[20];
  const float* noise_init  = (const float*)d_in[21];
  const float* noise_final = (const float*)d_in[22];

  float* out_slots = (float*)d_out;                 // [B,K,D]
  float* out_attn  = (float*)d_out + kB * kK * kD;  // [B,K,N,H]

  // workspace carve (256B aligned)
  char* w = (char*)d_ws;
  size_t off = 0;
  auto carve = [&](size_t bytes) { void* p = w + off; off += (bytes + 255) & ~(size_t)255; return p; };
  _Float16* xh    = (_Float16*)carve((size_t)kB * kN * kD * 2);
  _Float16* wkT   = (_Float16*)carve((size_t)kK * kD * kD * 2);
  _Float16* wvT   = (_Float16*)carve((size_t)kK * kD * kD * 2);
  _Float16* keys  = (_Float16*)carve((size_t)kB * kK * kN * kD * 2);
  _Float16* vals  = (_Float16*)carve((size_t)kB * kK * kN * kD * 2);
  float*    gll   = (float*)carve((size_t)kB * kK * kN * kH * 4);
  float*    slots = (float*)carve((size_t)kB * kK * kD * 4);
  float*    sigma = (float*)carve((size_t)kB * kK * kD * 4);
  float*    mixing= (float*)carve((size_t)kB * kK * 4);
  float*    qdiag = (float*)carve((size_t)kK * kD * 4);
  float*    accS0 = (float*)carve((size_t)kB * kK * kH * 4);          // 2048 B
  float*    accS1 = (float*)carve((size_t)kB * kK * kH * kDH * 4);    // 32768 B
  float*    accS2 = (float*)carve((size_t)kB * kK * kH * kDH * 4);
  float*    accV1 = (float*)carve((size_t)kB * kK * kH * kDH * 4);
  const size_t accBytes = 2048 + 3 * 32768;                           // contiguous (256B-padded sizes)

  init_kernel<<<32, 256, 0, stream>>>(slots_mu, slots_lsig, mixing_init, to_queries,
                                      noise_init, slots, sigma, mixing, qdiag);
  wconv_kernel<<<128, 256, 0, stream>>>(to_keys, to_values, wkT, wvT);
  ln_kernel<<<(kB * kN) / 8, 256, 0, stream>>>(embeddings, ln_in_g, ln_in_b, xh);
  gemm_kernel<<<dim3(kB * kK, kN / 64), 128, 0, stream>>>(xh, wkT, wvT, keys, vals);

  for (int it = 0; it < kITERS; ++it) {
    hipMemsetAsync(accS0, 0, accBytes, stream);
    gll_kernel<<<dim3(kB * kK, kN / 256), 256, 0, stream>>>(keys, slots, sigma, mixing,
                                                            qdiag, ln_sl_g, ln_sl_b, gll);
    attn_kernel<<<dim3(kN / 64, kB), 256, 0, stream>>>(gll, keys, vals,
                                                       accS0, accS1, accS2, accV1,
                                                       out_attn, (it == kITERS - 1) ? 1 : 0);
    finalize_kernel<<<kB * kK, 256, 0, stream>>>(accS0, accS1, accS2, accV1,
                                                 slots, sigma, mixing,
                                                 w_ih, w_hh, b_ih, b_hh,
                                                 w1, b1, w2, b2, ln_ff_g, ln_ff_b);
  }
  out_kernel<<<32, 256, 0, stream>>>(slots, sigma, noise_final, out_slots);
}
